// PSR4ConvF_58961311039788
// MI455X (gfx1250) — matrix operand order repaired and checked by reading, not compiled, not run on hardware
//
#include <hip/hip_runtime.h>

typedef __attribute__((ext_vector_type(16))) __bf16        bfrag_t;
typedef __attribute__((ext_vector_type(8)))  __bf16        v8bf;
typedef __attribute__((ext_vector_type(4)))  __bf16        v4bf;
typedef __attribute__((ext_vector_type(16))) float         v16f;
typedef __attribute__((ext_vector_type(8)))  float         v8f;
typedef __attribute__((ext_vector_type(4)))  float         v4f;
typedef __attribute__((ext_vector_type(4)))  unsigned int  v4u;

union Frag {
    bfrag_t        v;
    v8bf           b8[2];
    v4bf           b4[4];
    unsigned short us[16];
    v4u            q[2];
};

// ---------------------------------------------------------------------------
// Stage 1: za[(b,k,pos,o)] = relu( sum_c W[o,c,k] * x[b,c,pos] )  (bf16)
// GEMM: M = 576 (k*64+o), K = 64 (c), N = 32768 (b*4096+pos)
// One wave: 1 M-tile (16 rows) x 16 N-tiles.  36 Mtiles * 128 chunks = 4608 waves.
// ---------------------------------------------------------------------------
__global__ __launch_bounds__(256) void k1_zmix(const float* __restrict__ x,
                                               const float* __restrict__ W,
                                               unsigned short* __restrict__ za) {
    const int tid  = threadIdx.x;
    const int lane = tid & 31;
    const int wid  = tid >> 5;
    const int widx = blockIdx.x * 8 + wid;     // 0..4607
    const int mt    = widx % 36;               // M tile (16 rows of 576)
    const int chunk = widx / 36;               // 0..127, 16 N-tiles each
    const int half = lane >> 4;
    const int ln   = lane & 15;

    // A fragments: A(Mrow, c) = W[o, c, k],  Mrow = k*64 + o
    const int Mrow = mt * 16 + ln;
    const int oA   = Mrow & 63;
    const int kA   = Mrow >> 6;
    Frag a[2];
#pragma unroll
    for (int ki = 0; ki < 2; ++ki) {
        v8f lo, hi;
#pragma unroll
        for (int j = 0; j < 8; ++j) {
            const int c0 = ki * 32 + half * 8 + j;
            lo[j] = W[(oA * 64 + c0) * 9 + kA];
            hi[j] = W[(oA * 64 + c0 + 16) * 9 + kA];
        }
        a[ki].b8[0] = __builtin_convertvector(lo, v8bf);
        a[ki].b8[1] = __builtin_convertvector(hi, v8bf);
    }

    // store geometry: the 8 accum rows for this lane are o-contiguous
    const int Md0     = mt * 16 + 8 * half;    // first M row held by this lane
    const int kk      = Md0 >> 6;              // kernel position (tile-uniform)
    const int oo_base = Md0 & 63;              // multiple of 8 -> 16B aligned

    for (int t = chunk * 16; t < chunk * 16 + 16; ++t) {
        const int b   = t >> 8;                      // 256 tiles per batch image
        const int pos = ((t & 255) << 4) + ln;       // 0..4095
        v8f acc = {0.f, 0.f, 0.f, 0.f, 0.f, 0.f, 0.f, 0.f};
#pragma unroll
        for (int ki = 0; ki < 2; ++ki) {
            v16f xf;
#pragma unroll
            for (int j = 0; j < 16; ++j) {
                const int c = ki * 32 + half * 8 + (j & 7) + (j >> 3) * 16;
                xf[j] = x[(b * 64 + c) * 4096 + pos];
            }
            Frag bfr;
            bfr.v = __builtin_convertvector(xf, bfrag_t);
            acc = __builtin_amdgcn_wmma_f32_16x16x32_bf16(
                false, a[ki].v, false, bfr.v, (short)0, acc, false, false);
        }
        // relu + packed bf16 store (16 contiguous bytes per lane)
        v8f r;
#pragma unroll
        for (int v = 0; v < 8; ++v) r[v] = fmaxf(acc[v], 0.f);
        const v8bf rb = __builtin_convertvector(r, v8bf);
        *(v8bf*)(za + (size_t)(((b * 9 + kk) << 12) + pos) * 64 + oo_base) = rb;
    }
}

// ---------------------------------------------------------------------------
// Stage 2: out[b,o,rot,pos] = sum_{k,o'} W1[o, k*64+o'] * za[b,k,shift(pos),o']
// GEMM per (b,rot): M = 64, K = 576, N = 4096.
// One wave: 2 M-tiles (32 output channels, register-resident W1 frags)
//           x 8 spatial N-tiles.  2 * 32 * 32 = 2048 waves.
// ---------------------------------------------------------------------------
__global__ __launch_bounds__(256) void k2_gemm(const unsigned short* __restrict__ za,
                                               const float* __restrict__ W1,
                                               float* __restrict__ out) {
    const int tid  = threadIdx.x;
    const int lane = tid & 31;
    const int wid  = tid >> 5;
    const int widx = blockIdx.x * 8 + wid;   // 0..2047
    const int mg    = widx & 1;              // which 32 output channels
    const int br    = (widx >> 1) & 31;      // (b, rot)
    const int chunk = widx >> 6;             // 0..31, 8 spatial tiles each
    const int b   = br >> 2;
    const int rot = br & 3;
    const int half = lane >> 4;
    const int ln   = lane & 15;

    // Preload A fragments from W1 (f32 row-major, row stride 576)
    Frag a[18][2];
#pragma unroll
    for (int ki = 0; ki < 18; ++ki) {
#pragma unroll
        for (int s = 0; s < 2; ++s) {
            const int Mrow = mg * 32 + s * 16 + ln;
            const float* row = W1 + Mrow * 576;
            const int m0 = ki * 32 + half * 8;
            a[ki][s].b4[0] = __builtin_convertvector(*(const v4f*)(row + m0),      v4bf);
            a[ki][s].b4[1] = __builtin_convertvector(*(const v4f*)(row + m0 + 4),  v4bf);
            a[ki][s].b4[2] = __builtin_convertvector(*(const v4f*)(row + m0 + 16), v4bf);
            a[ki][s].b4[3] = __builtin_convertvector(*(const v4f*)(row + m0 + 20), v4bf);
        }
    }

    // Per-kstep ring shifts: RING dy=[-1,-1,-1,0,1,1,1,0], dx=[-1,0,1,1,1,0,-1,-1]
    int dyk[18], dxk[18];
#pragma unroll
    for (int ki = 0; ki < 18; ++ki) {
        const int k = ki >> 1;
        if (k == 8) { dyk[ki] = 0; dxk[ki] = 0; }
        else {
            const int r = (k + 2 * rot) & 7;
            dyk[ki] = ((0x6A40 >> (2 * r)) & 3) - 1;
            dxk[ki] = ((0x06A4 >> (2 * r)) & 3) - 1;
        }
    }

    const v4u zq = {0u, 0u, 0u, 0u};
    for (int t = chunk * 8; t < chunk * 8 + 8; ++t) {   // spatial tiles 0..255
        const int h = t >> 2;                 // 4 tiles per 64-wide row
        const int w = ((t & 3) << 4) + ln;
        v8f acc0 = {0.f, 0.f, 0.f, 0.f, 0.f, 0.f, 0.f, 0.f};
        v8f acc1 = {0.f, 0.f, 0.f, 0.f, 0.f, 0.f, 0.f, 0.f};
#pragma unroll
        for (int ki = 0; ki < 18; ++ki) {
            const int k  = ki >> 1;
            const int o0 = (ki & 1) * 32 + half * 8;
            const int hh = h + dyk[ki];
            const int ww = w + dxk[ki];
            const bool valid = ((unsigned)hh < 64u) && ((unsigned)ww < 64u);
            const int hc = valid ? hh : 0;
            const int wc = valid ? ww : 0;
            const unsigned short* p =
                za + (size_t)(((b * 9 + k) << 12) + (hc << 6) + wc) * 64 + o0;
            Frag bfr;
            const v4u q0 = *(const v4u*)p;
            const v4u q1 = *(const v4u*)(p + 16);
            bfr.q[0] = valid ? q0 : zq;
            bfr.q[1] = valid ? q1 : zq;
            acc0 = __builtin_amdgcn_wmma_f32_16x16x32_bf16(
                false, a[ki][0].v, false, bfr.v, (short)0, acc0, false, false);
            acc1 = __builtin_amdgcn_wmma_f32_16x16x32_bf16(
                false, a[ki][1].v, false, bfr.v, (short)0, acc1, false, false);
        }
        // store: out layout (B, 64, 4, 64, 64)
        const int pos = (t << 4) + ln;
#pragma unroll
        for (int v = 0; v < 8; ++v) {
            const int o_0 = mg * 32 + v + 8 * half;
            const int o_1 = o_0 + 16;
            out[(size_t)(((b * 64 + o_0) * 4 + rot) << 12) + pos] = acc0[v];
            out[(size_t)(((b * 64 + o_1) * 4 + rot) << 12) + pos] = acc1[v];
        }
    }
}

extern "C" void kernel_launch(void* const* d_in, const int* in_sizes, int n_in,
                              void* d_out, int out_size, void* d_ws, size_t ws_size,
                              hipStream_t stream) {
    const float* x  = (const float*)d_in[0];   // (8,64,64,64) f32
    const float* W  = (const float*)d_in[1];   // (64,64,9)    f32
    const float* W1 = (const float*)d_in[2];   // (64,576)     f32
    unsigned short* za = (unsigned short*)d_ws; // 8*9*4096*64 bf16 = 36 MB

    k1_zmix<<<576, 256, 0, stream>>>(x, W, za);              // 4608 waves exactly
    k2_gemm<<<256, 256, 0, stream>>>(za, W1, (float*)d_out); // 2048 waves exactly
}
